// ActQuantWrapper_82051055223253
// MI455X (gfx1250) — 8-bit matrix operand order repaired and checked by reading, compiled, not run on hardware
//
#include <hip/hip_runtime.h>
#include <stdint.h>

typedef int v8i __attribute__((ext_vector_type(8)));

#define D   4096
#define TOK 4096

__global__ __launch_bounds__(64)
void quant_weight_kernel(const float* __restrict__ w,
                         int8_t* __restrict__ qw,
                         float* __restrict__ sw) {
  const int row = blockIdx.x;
  const int t   = threadIdx.x;
  const float* wr = w + (size_t)row * D;

  float m = 0.0f;
  #pragma unroll 8
  for (int i = 0; i < 64; ++i) m = fmaxf(m, fabsf(wr[i * 64 + t]));

  __shared__ float red[64];
  red[t] = m;
  __syncthreads();
  #pragma unroll
  for (int s = 32; s > 0; s >>= 1) {
    if (t < s) red[t] = fmaxf(red[t], red[t + s]);
    __syncthreads();
  }
  const float scale = fmaxf(red[0] * (1.0f / 127.0f), 1e-5f);
  const float inv   = 1.0f / scale;
  if (t == 0) sw[row] = scale;

  const float4* src = (const float4*)(wr + t * 64);
  int4 packed[4];
  int* p = (int*)packed;
  #pragma unroll
  for (int i = 0; i < 16; ++i) {
    float4 v = src[i];
    int b0 = max(-127, min(127, (int)__builtin_rintf(v.x * inv)));
    int b1 = max(-127, min(127, (int)__builtin_rintf(v.y * inv)));
    int b2 = max(-127, min(127, (int)__builtin_rintf(v.z * inv)));
    int b3 = max(-127, min(127, (int)__builtin_rintf(v.w * inv)));
    p[i] = (b0 & 255) | ((b1 & 255) << 8) | ((b2 & 255) << 16) | ((b3 & 255) << 24);
  }
  int4* dst = (int4*)(qw + (size_t)row * D + t * 64);
  #pragma unroll
  for (int i = 0; i < 4; ++i) dst[i] = packed[i];
}

__global__ __launch_bounds__(64)
void had_quant_kernel(const float* __restrict__ x,
                      int8_t* __restrict__ qx,
                      float* __restrict__ sx) {
  const int token = blockIdx.x;
  const int d     = threadIdx.x;
  const float* xr = x + (size_t)token * D;

  float r[64];
  #pragma unroll
  for (int g = 0; g < 64; ++g) r[g] = xr[g * 64 + d];

  #pragma unroll
  for (int s = 1; s < 64; s <<= 1) {
    #pragma unroll
    for (int i = 0; i < 64; i += 2 * s) {
      #pragma unroll
      for (int j = 0; j < s; ++j) {
        float a = r[i + j];
        float b = r[i + j + s];
        r[i + j]     = a + b;
        r[i + j + s] = a - b;
      }
    }
  }

  float m = 0.0f;
  #pragma unroll
  for (int g = 0; g < 64; ++g) {
    r[g] *= 0.125f;
    m = fmaxf(m, fabsf(r[g]));
  }

  __shared__ float red[64];
  __shared__ __align__(16) int8_t qbuf[D];
  red[d] = m;
  __syncthreads();
  #pragma unroll
  for (int s = 32; s > 0; s >>= 1) {
    if (d < s) red[d] = fmaxf(red[d], red[d + s]);
    __syncthreads();
  }
  const float scale = fmaxf(red[0] * (1.0f / 127.0f), 1e-5f);
  const float inv   = 1.0f / scale;
  if (d == 0) sx[token] = scale;

  #pragma unroll
  for (int g = 0; g < 64; ++g) {
    int q = max(-127, min(127, (int)__builtin_rintf(r[g] * inv)));
    qbuf[g * 64 + d] = (int8_t)q;
  }
  __syncthreads();

  int4* dst = (int4*)(qx + (size_t)token * D + d * 64);
  const int4* src = (const int4*)(qbuf + d * 64);
  #pragma unroll
  for (int i = 0; i < 4; ++i) dst[i] = src[i];
}

__global__ __launch_bounds__(256)
void gemm_i8_kernel(const int8_t* __restrict__ qx,
                    const int8_t* __restrict__ qw,
                    const float* __restrict__ sx,
                    const float* __restrict__ sw,
                    const float* __restrict__ bias,
                    float* __restrict__ out) {
  const int bM   = blockIdx.y * 128;
  const int bN   = blockIdx.x * 128;
  const int tid  = threadIdx.x;
  const int lane = tid & 31;
  const int wave = tid >> 5;
  const int wm   = wave & 1;
  const int wn   = wave >> 1;
  const int l16  = lane & 15;
  const int half = lane >> 4;

  __shared__ __align__(16) int8_t At[128][64];
  __shared__ __align__(16) int8_t Bt[128][64];

  const v8i vzero = {0, 0, 0, 0, 0, 0, 0, 0};
  v8i acc[4][2];
  #pragma unroll
  for (int mi = 0; mi < 4; ++mi)
    #pragma unroll
    for (int ni = 0; ni < 2; ++ni) acc[mi][ni] = vzero;

  const int lrow = tid >> 1;
  const int lcol = (tid & 1) * 32;
  const int8_t* gA = qx + (size_t)(bM + lrow) * D + lcol;
  const int8_t* gB = qw + (size_t)(bN + lrow) * D + lcol;

  for (int kk = 0; kk < D; kk += 64) {
    const uint4* ga = (const uint4*)(gA + kk);
    const uint4* gb = (const uint4*)(gB + kk);
    *((uint4*)&At[lrow][lcol])      = ga[0];
    *((uint4*)&At[lrow][lcol + 16]) = ga[1];
    *((uint4*)&Bt[lrow][lcol])      = gb[0];
    *((uint4*)&Bt[lrow][lcol + 16]) = gb[1];
    __builtin_prefetch(gA + kk + 64, 0, 1);
    __builtin_prefetch(gB + kk + 64, 0, 1);
    __syncthreads();

    v8i bfrag[2];
    #pragma unroll
    for (int ni = 0; ni < 2; ++ni) {
      const int nr = wn * 32 + ni * 16 + l16;
      const uint2 b01 = *(const uint2*)&Bt[nr][half * 8];
      const uint2 b23 = *(const uint2*)&Bt[nr][16 + half * 8];
      const uint2 b45 = *(const uint2*)&Bt[nr][32 + half * 8];
      const uint2 b67 = *(const uint2*)&Bt[nr][48 + half * 8];
      v8i b;
      b[0] = (int)b01.x; b[1] = (int)b01.y; b[2] = (int)b23.x; b[3] = (int)b23.y;
      b[4] = (int)b45.x; b[5] = (int)b45.y; b[6] = (int)b67.x; b[7] = (int)b67.y;
      bfrag[ni] = b;
    }

    #pragma unroll
    for (int mi = 0; mi < 4; ++mi) {
      const int mr = wm * 64 + mi * 16 + l16;
      const uint2 a01 = *(const uint2*)&At[mr][half * 8];
      const uint2 a23 = *(const uint2*)&At[mr][16 + half * 8];
      const uint2 a45 = *(const uint2*)&At[mr][32 + half * 8];
      const uint2 a67 = *(const uint2*)&At[mr][48 + half * 8];
      v8i a;
      a[0] = (int)a01.x; a[1] = (int)a01.y; a[2] = (int)a23.x; a[3] = (int)a23.y;
      a[4] = (int)a45.x; a[5] = (int)a45.y; a[6] = (int)a67.x; a[7] = (int)a67.y;

      #pragma unroll
      for (int ni = 0; ni < 2; ++ni) {
        acc[mi][ni] = __builtin_amdgcn_wmma_i32_16x16x64_iu8(
            true, a, true, bfrag[ni], acc[mi][ni], false, false);
      }
    }
    __syncthreads();
  }

  const int mbase = half * 8;
  #pragma unroll
  for (int mi = 0; mi < 4; ++mi) {
    #pragma unroll
    for (int ni = 0; ni < 2; ++ni) {
      const int n   = bN + wn * 32 + ni * 16 + l16;
      const float s_n = sw[n];
      const float b_n = bias[n];
      #pragma unroll
      for (int r = 0; r < 8; ++r) {
        const int m = bM + wm * 64 + mi * 16 + mbase + r;
        out[(size_t)m * D + n] = (float)acc[mi][ni][r] * sx[m] * s_n + b_n;
      }
    }
  }
}

extern "C" void kernel_launch(void* const* d_in, const int* in_sizes, int n_in,
                              void* d_out, int out_size, void* d_ws, size_t ws_size,
                              hipStream_t stream) {
  const float* x    = (const float*)d_in[0];
  const float* w    = (const float*)d_in[1];
  const float* bias = (const float*)d_in[2];
  float* out = (float*)d_out;

  int8_t* qx = (int8_t*)d_ws;
  int8_t* qw = qx + (size_t)TOK * D;
  float*  sx = (float*)(qw + (size_t)D * D);
  float*  sw = sx + TOK;

  quant_weight_kernel<<<D, 64, 0, stream>>>(w, qw, sw);
  had_quant_kernel<<<TOK, 64, 0, stream>>>(x, qx, sx);
  gemm_i8_kernel<<<dim3(D / 128, TOK / 128), 256, 0, stream>>>(qx, qw, sx, sw, bias, out);
}
